// GCN_NET_69707319214250
// MI455X (gfx1250) — matrix operand order repaired and checked by reading, not compiled, not run on hardware
//
#include <hip/hip_runtime.h>

// ---------------------------------------------------------------------------
// GCNConv forward for MI455X (gfx1250, wave32, WMMA).
//   out = relu( D^-1/2 (A+I) D^-1/2 * (x @ W) + b )
// Pipeline (all on `stream`, graph-capture safe, no persistent state):
//   1) deg_init:   deg[i] = 1 (self loop)
//   2) deg_count:  deg[col[e]] += 1      (f32 atomics)
//   3) deg_fin:    deg[i] = rsqrt(deg[i])          -> dinv
//   4) pack_w:     W (f32) -> bf16 B-fragment layout (one 32KB blob, L2 forever)
//   5) gcn_gemm:   h_bf16 = bf16(x @ W) via v_wmma_f32_16x16x32_bf16
//                  and d_out = dinv^2 * h  (self-loop term, fresh overwrite)
//   6) scatter:    per edge: d_out[col] += dinv[row]*dinv[col] * h[row]
//                  32-edge batches, readlane (SGPR) broadcast, wave-wide
//                  256B bf16 gather + 4 f32 atomics/lane; h (25.6MB) and
//                  out (51.2MB) are L2-resident (192MB L2)
//   7) epilogue:   d_out = relu(d_out + b)  (in place)
// ---------------------------------------------------------------------------

typedef __attribute__((ext_vector_type(16))) __bf16 v16bf;
typedef __attribute__((ext_vector_type(8)))  float v8f;

union BF16x16 { unsigned int u[8]; v16bf v; };

__device__ __forceinline__ unsigned int f2bf(float f) {
    union { float f; unsigned int u; } x; x.f = f;
    unsigned int r = x.u + 0x7FFFu + ((x.u >> 16) & 1u);   // round-to-nearest-even
    return (r >> 16) & 0xFFFFu;
}
__device__ __forceinline__ float bf2f(unsigned int b) {
    union { unsigned int u; float f; } x; x.u = b << 16;
    return x.f;
}

// ---------------------------- degree kernels -------------------------------
__global__ void deg_init(float* __restrict__ deg, int n) {
    int i = blockIdx.x * blockDim.x + threadIdx.x;
    if (i < n) deg[i] = 1.0f;                       // self loop
}

__global__ void deg_count(const int* __restrict__ col, float* __restrict__ deg, int e) {
    int i = blockIdx.x * blockDim.x + threadIdx.x;
    if (i < e) atomicAdd(&deg[col[i]], 1.0f);
}

__global__ void deg_fin(float* __restrict__ deg, int n) {
    int i = blockIdx.x * blockDim.x + threadIdx.x;
    if (i < n) deg[i] = rsqrtf(deg[i]);             // deg >= 1 always
}

// ------------------- pack W into WMMA B-fragment layout --------------------
// B (32x16 bf16) fragment layout as packed here (contiguous k order, same as A):
//   lane 0-15 : n = lane,    K = 0..15  (VGPR v holds K=2v lo, K=2v+1 hi)
//   lane 16-31: n = lane-16, K = 16..31
// Wpack uint index: ((ct*4 + ks)*32 + lane)*8 + v   (ct = col tile, ks = K step)
__global__ void pack_w(const float* __restrict__ W, unsigned int* __restrict__ Wpack) {
    int t = blockIdx.x * blockDim.x + threadIdx.x;       // 8192 total
    if (t >= 8192) return;
    int v    = t & 7;
    int lane = (t >> 3) & 31;
    int ks   = (t >> 8) & 3;
    int ct   = t >> 10;
    int n  = ct * 16 + (lane & 15);
    int k0 = ks * 32 + ((lane >> 4) << 4) + (v << 1);
    unsigned int lo = f2bf(W[(size_t)k0 * 128 + n]);
    unsigned int hi = f2bf(W[(size_t)(k0 + 1) * 128 + n]);
    Wpack[t] = lo | (hi << 16);
}

// ------------------------------- GEMM --------------------------------------
// 256 threads = 8 waves. Block: 16 rows x 128 cols. Wave w: 16x16 tile, cols
// [16w, 16w+16). K=128 -> 4x v_wmma_f32_16x16x32_bf16.
// LDS x-tile row stride = 136 bf16 (=272B, 17 banks*16B) -> the two b128
// A-fragment loads are bank-conflict free (each half-wave hits all 64 banks).
__global__ __launch_bounds__(256) void gcn_gemm(
        const float* __restrict__ x, const unsigned int* __restrict__ Wpack,
        const float* __restrict__ dinv, unsigned short* __restrict__ h,
        float* __restrict__ out, int n) {
    __shared__ unsigned short xs[16 * 136];
    const int tid = threadIdx.x;
    const int rowBase = blockIdx.x << 4;

    // ---- stage 16x128 f32 -> bf16 LDS tile (each thread: 8 elements) ----
    {
        int r = tid >> 4;                 // 0..15
        int c = (tid & 15) << 3;          // 0,8,...,120
        int row = rowBase + r;
        if (row >= n) row = n - 1;        // safe clamp (stores are guarded)
        const float4* p = (const float4*)(x + (size_t)row * 128 + c);
        float4 a = p[0], b = p[1];
        uint4 st;
        st.x = f2bf(a.x) | (f2bf(a.y) << 16);
        st.y = f2bf(a.z) | (f2bf(a.w) << 16);
        st.z = f2bf(b.x) | (f2bf(b.y) << 16);
        st.w = f2bf(b.z) | (f2bf(b.w) << 16);
        *(uint4*)(&xs[r * 136 + c]) = st;
    }
    __syncthreads();

    const int wave = tid >> 5;
    const int lane = tid & 31;
    const int m    = lane & 15;
    const int half = lane >> 4;

    v8f acc = {};
    const unsigned int* wp = Wpack + ((size_t)(wave * 128 + lane) << 3);

#pragma unroll
    for (int ks = 0; ks < 4; ++ks) {
        // A fragment (16x32 bf16): two b128 LDS reads
        //   V0-3: A[m][ks*32 + half*16 + 0..7], V4-7: +8
        const unsigned short* ap = &xs[m * 136 + ks * 32 + half * 16];
        uint4 alo = *(const uint4*)(ap);
        uint4 ahi = *(const uint4*)(ap + 8);
        BF16x16 a;
        a.u[0] = alo.x; a.u[1] = alo.y; a.u[2] = alo.z; a.u[3] = alo.w;
        a.u[4] = ahi.x; a.u[5] = ahi.y; a.u[6] = ahi.z; a.u[7] = ahi.w;
        // B fragment: 8 consecutive dwords per lane, fully coalesced
        const uint4* bp = (const uint4*)(wp + ks * 256);
        uint4 blo = bp[0];
        uint4 bhi = bp[1];
        BF16x16 b;
        b.u[0] = blo.x; b.u[1] = blo.y; b.u[2] = blo.z; b.u[3] = blo.w;
        b.u[4] = bhi.x; b.u[5] = bhi.y; b.u[6] = bhi.z; b.u[7] = bhi.w;

        acc = __builtin_amdgcn_wmma_f32_16x16x32_bf16(
                false, a.v, false, b.v, (short)0, acc, false, false);
    }

    // ---- epilogue: h (bf16) + self-loop term into d_out (fresh write) ----
    // D layout: lane n = lane&15; VGPR v -> row m = v + 8*(lane>>4)
    // Uniform fast path: only the last block can be a partial tile.
    const int colIdx = (wave << 4) + m;
    const int rowOff = rowBase + (half << 3);
    if (rowBase + 16 <= n) {
#pragma unroll
        for (int v = 0; v < 8; ++v) {
            int row = rowOff + v;
            float val = acc[v];
            size_t idx = (size_t)row * 128 + colIdx;
            h[idx] = (unsigned short)f2bf(val);
            float di = dinv[row];
            out[idx] = di * di * val;
        }
    } else {
#pragma unroll
        for (int v = 0; v < 8; ++v) {
            int row = rowOff + v;
            if (row < n) {
                float val = acc[v];
                size_t idx = (size_t)row * 128 + colIdx;
                h[idx] = (unsigned short)f2bf(val);
                float di = dinv[row];
                out[idx] = di * di * val;
            }
        }
    }
}

// ------------------------------ scatter ------------------------------------
// 32-edge batches per wave. Each lane loads its own edge's (row,col,norm)
// with coalesced loads, then 32 unrolled iterations broadcast one edge via
// v_readlane (SGPR) -> gather/atomic addressing is SGPR-base + lane offset:
//   global_load_b64  v, saddr   (256B bf16 row gather, L2-resident)
//   global_atomic_add_f32 x4    (512B scatter into L2-resident out)
// Tail lanes use norm=0 (value-preserving +0.0 adds), keeping the loop
// branch-free and fully unrolled.
__global__ __launch_bounds__(256) void scatter(
        const int* __restrict__ row, const int* __restrict__ col,
        const float* __restrict__ dinv, const unsigned short* __restrict__ h,
        float* __restrict__ out, int E) {
    const int lane = threadIdx.x & 31;
    const int wid  = (blockIdx.x * blockDim.x + threadIdx.x) >> 5;
    const int nw   = (gridDim.x * blockDim.x) >> 5;
    const int stride = nw << 5;

    for (int base = wid << 5; base < E; base += stride) {
        const int e = base + lane;
        int r = 0, c = 0;
        float nrm = 0.0f;
        if (e < E) {
            r = row[e];
            c = col[e];
            nrm = dinv[r] * dinv[c];
        }
#pragma unroll
        for (int j = 0; j < 32; ++j) {
            const int   rj = __builtin_amdgcn_readlane(r, j);
            const int   cj = __builtin_amdgcn_readlane(c, j);
            const float nj = __uint_as_float(
                __builtin_amdgcn_readlane(__float_as_uint(nrm), j));
            const uint2 hv = *(const uint2*)(h + (size_t)rj * 128 + (lane << 2));
            float m0 = bf2f(hv.x & 0xFFFFu) * nj;
            float m1 = bf2f(hv.x >> 16)     * nj;
            float m2 = bf2f(hv.y & 0xFFFFu) * nj;
            float m3 = bf2f(hv.y >> 16)     * nj;
            float* o = out + (size_t)cj * 128 + (lane << 2);
            atomicAdd(o + 0, m0);
            atomicAdd(o + 1, m1);
            atomicAdd(o + 2, m2);
            atomicAdd(o + 3, m3);
        }
    }
}

// ------------------------------ epilogue -----------------------------------
__global__ void epilogue(float* __restrict__ out, const float* __restrict__ b, int total4) {
    int i = blockIdx.x * blockDim.x + threadIdx.x;
    if (i >= total4) return;
    float4 v  = ((float4*)out)[i];
    float4 bb = ((const float4*)b)[i & 31];          // 128 cols = 32 float4
    v.x = fmaxf(v.x + bb.x, 0.0f);
    v.y = fmaxf(v.y + bb.y, 0.0f);
    v.z = fmaxf(v.z + bb.z, 0.0f);
    v.w = fmaxf(v.w + bb.w, 0.0f);
    ((float4*)out)[i] = v;
}

// ---------------------------------------------------------------------------
extern "C" void kernel_launch(void* const* d_in, const int* in_sizes, int n_in,
                              void* d_out, int out_size, void* d_ws, size_t ws_size,
                              hipStream_t stream) {
    const float* x  = (const float*)d_in[0];
    const int*   ei = (const int*)d_in[1];      // [2,E] int (row ; col)
    const float* W  = (const float*)d_in[2];
    const float* b  = (const float*)d_in[3];
    float* out = (float*)d_out;

    const int N = in_sizes[0] / 128;
    const int E = in_sizes[1] / 2;
    const int* rowIdx = ei;
    const int* colIdx = ei + E;

    // workspace: deg/dinv (N f32) | Wpack (8192 u32) | h (N*128 bf16) ~26MB
    char* ws = (char*)d_ws;
    float* deg = (float*)ws;
    size_t off = ((size_t)N * 4 + 255) & ~(size_t)255;
    unsigned int* Wpack = (unsigned int*)(ws + off);
    off += 8192 * 4;
    unsigned short* h = (unsigned short*)(ws + off);

    deg_init<<<(N + 255) / 256, 256, 0, stream>>>(deg, N);
    deg_count<<<(E + 255) / 256, 256, 0, stream>>>(colIdx, deg, E);
    deg_fin<<<(N + 255) / 256, 256, 0, stream>>>(deg, N);
    pack_w<<<32, 256, 0, stream>>>(W, Wpack);
    gcn_gemm<<<(N + 15) / 16, 256, 0, stream>>>(x, Wpack, deg, h, out, N);
    scatter<<<2048, 256, 0, stream>>>(rowIdx, colIdx, deg, h, out, E);
    epilogue<<<(N * 128 / 4 + 255) / 256, 256, 0, stream>>>(out, b, N * 128 / 4);
}
